// KptPrototype_55834574848218
// MI455X (gfx1250) — matrix operand order repaired and checked by reading, not compiled, not run on hardware
//
#include <hip/hip_runtime.h>
#include <cstdint>
#include <cstddef>

#define NKPT 17
#define MPR  10
#define DIMF 768
#define NPTS 131072
#define KM   170            // NKPT*MPR
#define KMP  176            // padded to 11 tiles of 16 columns
#define EPSV   0.05f
#define GAMMAV 0.999f
#define CLAMPV 0.0001f

typedef __attribute__((ext_vector_type(16))) __bf16 v16bf;
typedef __attribute__((ext_vector_type(8)))  float  v8f;

__device__ __forceinline__ uint32_t hash32(uint32_t x){
  x ^= x >> 16; x *= 0x7feb352du; x ^= x >> 15; x *= 0x846ca68bu; x ^= x >> 16; return x;
}
__device__ __forceinline__ unsigned short bf16bits(float f){
  __bf16 h = (__bf16)f; return __builtin_bit_cast(unsigned short, h);
}

// ---------------------------------------------------------------- utilities
__global__ __launch_bounds__(256) void zero_kernel(float* p, int count){
  int i = blockIdx.x*256 + threadIdx.x;
  if (i < count) p[i] = 0.0f;
}

// per-row inverse L2 norm of feats (one wave32 per row)
__global__ __launch_bounds__(256) void rnorm_kernel(const float* __restrict__ feats,
                                                    float* __restrict__ rnorm){
  int lane = threadIdx.x & 31, wave = threadIdx.x >> 5;
  int row  = blockIdx.x*8 + wave;
  const float* fp = feats + (size_t)row*DIMF;
  float s = 0.f;
  for (int d = lane; d < DIMF; d += 32){ float v = fp[d]; s += v*v; }
  #pragma unroll
  for (int off = 16; off > 0; off >>= 1) s += __shfl_xor(s, off, 32);
  if (lane == 0) rnorm[row] = 1.0f / fmaxf(sqrtf(s), 1e-12f);
}

// normalize prototypes -> f32 (for EMA update) and bf16 B matrix (padded rows zeroed)
__global__ __launch_bounds__(256) void protoprep_kernel(const float* __restrict__ protos,
                                                        float* __restrict__ protosN,
                                                        unsigned short* __restrict__ protosB){
  __shared__ float red[256];
  int c = blockIdx.x, t = threadIdx.x;
  if (c >= KM){
    for (int d = t; d < DIMF; d += 256) protosB[(size_t)c*DIMF + d] = 0;
    return;
  }
  const float* src = protos + (size_t)c*DIMF;
  float v0 = src[t], v1 = src[t+256], v2 = src[t+512];
  red[t] = v0*v0 + v1*v1 + v2*v2; __syncthreads();
  for (int o = 128; o > 0; o >>= 1){ if (t < o) red[t] += red[t+o]; __syncthreads(); }
  float inv = 1.0f / fmaxf(sqrtf(red[0]), 1e-12f);
  float n0 = v0*inv, n1 = v1*inv, n2 = v2*inv;
  protosN[(size_t)c*DIMF + t      ] = n0;
  protosN[(size_t)c*DIMF + t + 256] = n1;
  protosN[(size_t)c*DIMF + t + 512] = n2;
  protosB[(size_t)c*DIMF + t      ] = bf16bits(n0);
  protosB[(size_t)c*DIMF + t + 256] = bf16bits(n1);
  protosB[(size_t)c*DIMF + t + 512] = bf16bits(n2);
}

// ---------------------------------------------------------------- main GEMM
// C(N x 176) = l2n(feats)(N x 768, bf16 on the fly) @ protosB^T, abs() stored.
// Block = 8 waves, wave w -> 16 rows; 11 column tiles of 16; K stepped by 32.
// B slices (176 x 32 bf16 = 11 KB) double-buffered in LDS via CDNA5 async LDS
// loads so the DMA overlaps the WMMA chain. sched_barrier fences force all 22
// ds_load_b128 to issue before the WMMA chain (one dscnt wait, 11 WMMAs
// back-to-back with no memory deps).
__global__ __launch_bounds__(256, 1) void gemm_kernel(const float* __restrict__ feats,
                                                      const float* __restrict__ rnorm,
                                                      const unsigned short* __restrict__ protosB,
                                                      float* __restrict__ logits){
  __shared__ __align__(16) unsigned short Bs[2][KMP*32];
  const int lane = threadIdx.x & 31;
  const int wave = threadIdx.x >> 5;
  const int half = lane >> 4;           // 0: K 0..7 / 16..23   1: K 8..15 / 24..31
  const int l16  = lane & 15;
  const int rowBase = blockIdx.x*128 + wave*16;
  const int row  = rowBase + l16;       // A row held by this lane (both halves)
  const float rn = rnorm[row];
  const uint64_t bbase = (uint64_t)(uintptr_t)protosB;

  auto stage = [&](int kb, int buf){
    for (int i = threadIdx.x; i < KMP*4; i += 256){
      int c  = i >> 2;
      int k8 = (i & 3) * 8;                       // 8 bf16 = 16 bytes per transfer
      unsigned ldsOff = (unsigned)(uintptr_t)(&Bs[buf][c*32 + k8]);
      unsigned gOff   = (unsigned)((c*DIMF + kb + k8) * 2);
      asm volatile("global_load_async_to_lds_b128 %0, %1, %2 offset:0"
                   :: "v"(ldsOff), "v"(gOff), "s"(bbase) : "memory");
    }
  };

  v8f acc[11];
  #pragma unroll
  for (int t = 0; t < 11; ++t)
    #pragma unroll
    for (int j = 0; j < 8; ++j) acc[t][j] = 0.0f;

  // prologue: stage first slice
  stage(0, 0);
  asm volatile("s_wait_asynccnt 0" ::: "memory");
  __syncthreads();

  int cur = 0;
  for (int kb = 0; kb < DIMF; kb += 32){
    // kick off async DMA of the next slice into the other buffer (overlaps compute)
    if (kb + 32 < DIMF) stage(kb + 32, cur ^ 1);

    // ---- A fragment: 16 bf16 per lane per ISA layout, f32->bf16 in registers
    const float* ap = feats + (size_t)row*DIMF + kb + half*8;
    float4 x0 = *(const float4*)(ap);
    float4 x1 = *(const float4*)(ap + 4);
    float4 y0 = *(const float4*)(ap + 16);
    float4 y1 = *(const float4*)(ap + 20);
    v16bf a;
    a[0]=(__bf16)(x0.x*rn); a[1]=(__bf16)(x0.y*rn); a[2]=(__bf16)(x0.z*rn); a[3]=(__bf16)(x0.w*rn);
    a[4]=(__bf16)(x1.x*rn); a[5]=(__bf16)(x1.y*rn); a[6]=(__bf16)(x1.z*rn); a[7]=(__bf16)(x1.w*rn);
    a[8]=(__bf16)(y0.x*rn); a[9]=(__bf16)(y0.y*rn); a[10]=(__bf16)(y0.z*rn); a[11]=(__bf16)(y0.w*rn);
    a[12]=(__bf16)(y1.x*rn); a[13]=(__bf16)(y1.y*rn); a[14]=(__bf16)(y1.z*rn); a[15]=(__bf16)(y1.w*rn);

    // ---- preload ALL 11 B fragments; fence so the scheduler cannot sink the
    // ds_loads into the WMMA chain (one dscnt wait instead of 11)
    __builtin_amdgcn_sched_barrier(0);
    uint4 bu[11][2];
    #pragma unroll
    for (int ct = 0; ct < 11; ++ct){
      const uint4* bp = (const uint4*)(&Bs[cur][(ct*16 + l16)*32 + half*8]);
      bu[ct][0] = bp[0];
      bu[ct][1] = bp[2];
    }
    __builtin_amdgcn_sched_barrier(0);

    // ---- 11 WMMAs back-to-back, no memory deps inside the chain
    #pragma unroll
    for (int ct = 0; ct < 11; ++ct){
      union { uint4 u[2]; v16bf v; } bb;
      bb.u[0] = bu[ct][0]; bb.u[1] = bu[ct][1];
      acc[ct] = __builtin_amdgcn_wmma_f32_16x16x32_bf16(
          false, a, false, bb.v, (short)0, acc[ct], false, false);
    }
    __builtin_amdgcn_sched_barrier(0);

    // next buffer ready + everyone done reading cur before it is overwritten
    asm volatile("s_wait_asynccnt 0" ::: "memory");
    __syncthreads();
    cur ^= 1;
  }

  // D layout: lane column = l16, rows = v + 8*half
  #pragma unroll
  for (int ct = 0; ct < 11; ++ct){
    int c = ct*16 + l16;
    if (c < KM){
      #pragma unroll
      for (int v = 0; v < 8; ++v){
        int r = rowBase + v + half*8;
        logits[(size_t)r*KM + c] = fabsf(acc[ct][v]);
      }
    }
  }
}

// ---------------------------------------------------------------- epilogue passes
__global__ __launch_bounds__(256) void post_kernel(const float* __restrict__ logits,
      const int* __restrict__ gt, const float* __restrict__ conf,
      float* __restrict__ kcls, float* __restrict__ conf_eff, float* __restrict__ L,
      float* __restrict__ Ssum, float* __restrict__ cnt){
  int n = blockIdx.x*256 + threadIdx.x;
  const float* lg = logits + (size_t)n*KM;
  int g = gt[n];
  float best = -1.f; int pred = 0;
  for (int k = 0; k < NKPT; ++k){
    float mx = 0.f;
    for (int m = 0; m < MPR; ++m) mx = fmaxf(mx, lg[k*MPR + m]);
    kcls[(size_t)n*NKPT + k] = fminf(fmaxf(mx, CLAMPV), 1.0f - CLAMPV);
    if (mx > best){ best = mx; pred = k; }
  }
  conf_eff[n] = (g == pred) ? conf[n] : 0.f;
  float rs = 0.f;
  for (int m = 0; m < MPR; ++m){
    float l = __expf(lg[g*MPR + m] / EPSV);
    L[(size_t)n*MPR + m] = l; rs += l;
  }
  atomicAdd(&Ssum[g], rs);
  atomicAdd(&cnt[g], 1.0f);
}

__global__ __launch_bounds__(256) void scaleL_kernel(const int* __restrict__ gt,
      const float* __restrict__ Ssum, float* __restrict__ L){
  int n = blockIdx.x*256 + threadIdx.x;
  float inv = 1.0f / Ssum[gt[n]];
  for (int m = 0; m < MPR; ++m) L[(size_t)n*MPR + m] *= inv;
}

__global__ __launch_bounds__(256) void colsum_kernel(const int* __restrict__ gt,
      const float* __restrict__ L, float* __restrict__ colsum){
  int n = blockIdx.x*256 + threadIdx.x;
  int g = gt[n];
  for (int m = 0; m < MPR; ++m) atomicAdd(&colsum[g*MPR + m], L[(size_t)n*MPR + m]);
}

__global__ __launch_bounds__(256) void rownorm_kernel(const int* __restrict__ gt,
      const float* __restrict__ cnt, const float* __restrict__ colsum, float* __restrict__ L){
  int n = blockIdx.x*256 + threadIdx.x;
  int g = gt[n];
  float B = cnt[g];
  float l[MPR]; float rs = 0.f;
  for (int m = 0; m < MPR; ++m){
    l[m] = L[(size_t)n*MPR + m] / (colsum[g*MPR + m] * (float)MPR);
    rs += l[m];
  }
  float inv = (rs > 0.f) ? 1.0f/(rs*B) : 0.f;
  for (int m = 0; m < MPR; ++m) L[(size_t)n*MPR + m] = l[m]*inv;
}

__global__ __launch_bounds__(256) void finalize_kernel(const int* __restrict__ gt,
      const float* __restrict__ cnt, const float* __restrict__ conf_eff,
      const float* __restrict__ L, float* __restrict__ out_target,
      int* __restrict__ assign, float* __restrict__ wsq, float* __restrict__ nsum){
  int n = blockIdx.x*256 + threadIdx.x;
  int g = gt[n];
  float B = cnt[g];
  float mx = -1e30f; int idx = 0;
  float my = -1e30f; int h = 0;
  for (int m = 0; m < MPR; ++m){
    float l = L[(size_t)n*MPR + m] * B;
    if (l > mx){ mx = l; idx = m; }
    uint32_t r = hash32((uint32_t)(n*MPR + m)*0x9E3779B9u + 42u);
    float u = fmaxf((r >> 8) * (1.0f/16777216.0f), 1e-7f);
    float gmb = -logf(fmaxf(-logf(u), 1e-20f));
    float y = l + gmb;                       // argmax of softmax((L+g)/tau)
    if (y > my){ my = y; h = m; }
  }
  out_target[n] = (float)(idx + MPR*g);
  assign[n] = h;
  float w = conf_eff[n];
  wsq[n] = w*w;
  if (w > 0.f) atomicAdd(&nsum[g*MPR + h], w);
}

// f[k,m,:] += conf_eff^2 * l2n(feats[n,:]) scattered by (gt, hard assignment)
__global__ __launch_bounds__(256) void faccum_kernel(const float* __restrict__ feats,
      const float* __restrict__ rnorm, const int* __restrict__ gt,
      const int* __restrict__ assign, const float* __restrict__ wsq,
      float* __restrict__ facc){
  int n = blockIdx.x;
  float w = wsq[n];
  if (w == 0.f) return;
  float rn = rnorm[n];
  const float* fp = feats + (size_t)n*DIMF;
  float* dst = facc + ((size_t)gt[n]*MPR + assign[n])*DIMF;
  for (int d = threadIdx.x; d < DIMF; d += 256)
    atomicAdd(&dst[d], w * fp[d] * rn);
}

__global__ __launch_bounds__(256) void update_kernel(const float* __restrict__ protosN,
      const float* __restrict__ facc, const float* __restrict__ nsum,
      float* __restrict__ outp){
  __shared__ float red[256];
  int j = blockIdx.x, t = threadIdx.x;
  int k = j / MPR;
  float hs = 0.f;
  for (int mm = 0; mm < MPR; ++mm) hs += nsum[k*MPR + mm];
  bool active = (hs > 0.f) && (nsum[j] != 0.f);
  const float* fr = facc    + (size_t)j*DIMF;
  const float* pr = protosN + (size_t)j*DIMF;
  float f0 = fr[t], f1 = fr[t+256], f2 = fr[t+512];
  red[t] = f0*f0 + f1*f1 + f2*f2; __syncthreads();
  for (int o = 128; o > 0; o >>= 1){ if (t < o) red[t] += red[t+o]; __syncthreads(); }
  float finv = 1.0f / fmaxf(sqrtf(red[0]), 1e-12f);
  __syncthreads();
  float p0 = pr[t], p1 = pr[t+256], p2 = pr[t+512];
  float n0, n1, n2;
  if (active){
    n0 = GAMMAV*p0 + (1.0f-GAMMAV)*f0*finv;
    n1 = GAMMAV*p1 + (1.0f-GAMMAV)*f1*finv;
    n2 = GAMMAV*p2 + (1.0f-GAMMAV)*f2*finv;
  } else { n0 = p0; n1 = p1; n2 = p2; }
  red[t] = n0*n0 + n1*n1 + n2*n2; __syncthreads();
  for (int o = 128; o > 0; o >>= 1){ if (t < o) red[t] += red[t+o]; __syncthreads(); }
  float inv = 1.0f / fmaxf(sqrtf(red[0]), 1e-12f);
  outp[(size_t)j*DIMF + t      ] = n0*inv;
  outp[(size_t)j*DIMF + t + 256] = n1*inv;
  outp[(size_t)j*DIMF + t + 512] = n2*inv;
}

// ---------------------------------------------------------------- launch
extern "C" void kernel_launch(void* const* d_in, const int* in_sizes, int n_in,
                              void* d_out, int out_size, void* d_ws, size_t ws_size,
                              hipStream_t stream){
  const float* feats  = (const float*)d_in[0];
  const int*   gt     = (const int*)d_in[1];
  const float* conf   = (const float*)d_in[2];
  const float* protos = (const float*)d_in[3];

  char* w = (char*)d_ws;
  auto carve = [&](size_t bytes) -> void* {
    void* p = (void*)w; w += (bytes + 255) & ~(size_t)255; return p;
  };
  float* rnorm    = (float*)carve((size_t)NPTS*4);
  float* conf_eff = (float*)carve((size_t)NPTS*4);
  float* wsq      = (float*)carve((size_t)NPTS*4);
  int*   assign   = (int*)  carve((size_t)NPTS*4);
  float* L        = (float*)carve((size_t)NPTS*MPR*4);
  float* protosN  = (float*)carve((size_t)KM*DIMF*4);
  unsigned short* protosB = (unsigned short*)carve((size_t)KMP*DIMF*2);
  const int ZCOUNT = NKPT + NKPT + 3*KM + KM + KM*DIMF;
  float* zbase  = (float*)carve((size_t)ZCOUNT*4);
  float* Ssum   = zbase;
  float* cnt    = Ssum + NKPT;
  float* colsum = cnt + NKPT;          // 3 * KM (one buffer per Sinkhorn iter)
  float* nsum   = colsum + 3*KM;
  float* facc   = nsum + KM;

  float* out        = (float*)d_out;
  float* out_logits = out;                                  // N*KM
  float* out_target = out_logits + (size_t)NPTS*KM;         // N
  float* out_kcls   = out_target + NPTS;                    // N*NKPT
  float* out_nproto = out_kcls + (size_t)NPTS*NKPT;         // KM*DIMF

  zero_kernel    <<<(ZCOUNT+255)/256, 256, 0, stream>>>(zbase, ZCOUNT);
  rnorm_kernel   <<<NPTS/8,   256, 0, stream>>>(feats, rnorm);
  protoprep_kernel<<<KMP,     256, 0, stream>>>(protos, protosN, protosB);
  gemm_kernel    <<<NPTS/128, 256, 0, stream>>>(feats, rnorm, protosB, out_logits);
  post_kernel    <<<NPTS/256, 256, 0, stream>>>(out_logits, gt, conf, out_kcls,
                                                conf_eff, L, Ssum, cnt);
  scaleL_kernel  <<<NPTS/256, 256, 0, stream>>>(gt, Ssum, L);
  for (int it = 0; it < 3; ++it){
    colsum_kernel <<<NPTS/256, 256, 0, stream>>>(gt, L, colsum + it*KM);
    rownorm_kernel<<<NPTS/256, 256, 0, stream>>>(gt, cnt, colsum + it*KM, L);
  }
  finalize_kernel<<<NPTS/256, 256, 0, stream>>>(gt, cnt, conf_eff, L,
                                                out_target, assign, wsq, nsum);
  faccum_kernel  <<<NPTS,     256, 0, stream>>>(feats, rnorm, gt, assign, wsq, facc);
  update_kernel  <<<KM,       256, 0, stream>>>(protosN, facc, nsum, out_nproto);
}
